// CrossViewSwapAttention_75282186764523
// MI455X (gfx1250) — matrix operand order repaired and checked by reading, not compiled, not run on hardware
//
#include <hip/hip_runtime.h>
#include <cstdint>

// ---------------------------------------------------------------------------
// CDNA5 (gfx1250) CrossViewSwapAttention pipeline, f16-WMMA for all GEMMs.
// wave32, 256-thread (8-wave) blocks throughout.
// GEMM stages tiles with gfx1250 async global->LDS DMA (ASYNCcnt) and
// double-buffers LDS so WMMA overlaps the next tile's staging.
// ---------------------------------------------------------------------------

typedef _Float16 half_t;
typedef __attribute__((ext_vector_type(16))) _Float16 v16h;
typedef __attribute__((ext_vector_type(8)))  float    v8f;

#define BN_S 0.9999950000374996f

union Frag { v16h h; unsigned int u[8]; };

// A fragment (16x32 f16): lane L holds row m=L&15; khalf=(L>>4):
//   VGPR v<4 : K = khalf*8 + 2v,2v+1 ; VGPR v>=4 : K = 16 + khalf*8 + 2(v-4)..
__device__ inline v16h load_fragA(const unsigned int* rowp, int colHalves, int lane) {
    const int khalf = (lane >> 4) & 1;
    const int base = (colHalves >> 1) + khalf * 4;
    Frag f;
#pragma unroll
    for (int v = 0; v < 4; ++v) f.u[v] = rowp[base + v];
#pragma unroll
    for (int v = 0; v < 4; ++v) f.u[4 + v] = rowp[base + 8 + v];
    return f.h;
}

// B fragment (32x16 f16): lane L holds column n=L&15; khalf=(L>>4), same K
// order as A: v<4 : K = khalf*8 + 2v.. ; v>=4 : K = 16 + khalf*8 + 2(v-4)..
// rowp points at row n of [N,K] storage.
__device__ inline v16h load_fragB(const unsigned int* rowp, int colHalves, int lane) {
    const int base = (colHalves >> 1) + ((lane >> 4) & 1) * 4;
    Frag f;
#pragma unroll
    for (int v = 0; v < 4; ++v) f.u[v] = rowp[base + v];
#pragma unroll
    for (int v = 0; v < 4; ++v) f.u[4 + v] = rowp[base + 8 + v];
    return f.h;
}

// gfx1250 async global -> LDS DMA, 16B per lane. Tracked by ASYNCcnt.
// VDST = LDS byte address (low 32 bits of generic shared pointer), VADDR = VA.
__device__ inline void async_g2l_b128(void* ldsPtr, const void* gptr) {
    const unsigned ldsAddr = (unsigned)(uintptr_t)ldsPtr;
    asm volatile("global_load_async_to_lds_b128 %0, %1, off"
                 :: "v"(ldsAddr), "v"(gptr)
                 : "memory");
}
__device__ inline void wait_asynccnt0() {
    asm volatile("s_wait_asynccnt 0" ::: "memory");
}

__device__ inline float gelu_f(float x) {
    return 0.5f * x * (1.0f + erff(x * 0.70710678118654752f));
}

__device__ inline float wave_sum(float v) {
#pragma unroll
    for (int m = 16; m >= 1; m >>= 1) v += __shfl_xor(v, m, 32);
    return v;
}

// ---------------------------------------------------------------------------
// Generic f16 WMMA GEMM:  C[M,N] = act( preScale * A[M,K] @ Bw[N,K]^T + bias ) + resid
// Block tile 64(M) x 128(N); K in steps of 32. M%64==0, N%128==0, K%32==0.
// Double-buffered async LDS staging.
// ---------------------------------------------------------------------------
__global__ __launch_bounds__(256)
void k_gemm(const half_t* __restrict__ A, const half_t* __restrict__ Bw,
            const float* __restrict__ bias, const float* __restrict__ resid,
            float* __restrict__ outF, half_t* __restrict__ outH,
            int M, int N, int K, float preScale, int actGelu, int storeT)
{
    __shared__ half_t As[2][64][40];
    __shared__ half_t Bs[2][128][40];
    const int t = threadIdx.x;
    const int r0 = blockIdx.x * 64;
    const int c0 = blockIdx.y * 128;
    const int w = t >> 5, lane = t & 31;
    const int wm = w >> 1, wn = w & 1;

    v8f acc[4];
#pragma unroll
    for (int j = 0; j < 4; ++j)
#pragma unroll
        for (int i = 0; i < 8; ++i) acc[j][i] = 0.0f;

    const int aRow = r0 + (t >> 2);
    const int aCol = (t & 3) * 8;
    const int bRow = c0 + (t >> 1);
    const int bCol = (t & 1) * 16;

    // stage first K-tile into buffer 0
    async_g2l_b128(&As[0][t >> 2][aCol],     A  + (size_t)aRow * K + aCol);
    async_g2l_b128(&Bs[0][t >> 1][bCol],     Bw + (size_t)bRow * K + bCol);
    async_g2l_b128(&Bs[0][t >> 1][bCol + 8], Bw + (size_t)bRow * K + bCol + 8);

    for (int kt = 0; kt < K; kt += 32) {
        const int cur = (kt >> 5) & 1;
        wait_asynccnt0();          // our async writes to LDS done
        __syncthreads();           // everyone's writes visible; prior reads done
        if (kt + 32 < K) {         // overlap: stream next tile while computing
            const int nxt = cur ^ 1;
            async_g2l_b128(&As[nxt][t >> 2][aCol],     A  + (size_t)aRow * K + kt + 32 + aCol);
            async_g2l_b128(&Bs[nxt][t >> 1][bCol],     Bw + (size_t)bRow * K + kt + 32 + bCol);
            async_g2l_b128(&Bs[nxt][t >> 1][bCol + 8], Bw + (size_t)bRow * K + kt + 32 + bCol + 8);
            if (kt + 64 < K) {
                __builtin_prefetch(A  + (size_t)aRow * K + kt + 64 + aCol, 0, 1);
                __builtin_prefetch(Bw + (size_t)bRow * K + kt + 64 + bCol, 0, 1);
            }
        }
        const unsigned int* arow = (const unsigned int*)(&As[cur][wm * 16 + (lane & 15)][0]);
        v16h af = load_fragA(arow, 0, lane);
#pragma unroll
        for (int j = 0; j < 4; ++j) {
            const unsigned int* brow =
                (const unsigned int*)(&Bs[cur][wn * 64 + j * 16 + (lane & 15)][0]);
            v16h bf = load_fragB(brow, 0, lane);
            acc[j] = __builtin_amdgcn_wmma_f32_16x16x32_f16(false, af, false, bf,
                                                            (short)0, acc[j], false, false);
        }
    }

#pragma unroll
    for (int j = 0; j < 4; ++j) {
#pragma unroll
        for (int i = 0; i < 8; ++i) {
            const int m = wm * 16 + ((lane >> 4) & 1) * 8 + i;
            const int n = wn * 64 + j * 16 + (lane & 15);
            const int gr = r0 + m, gc = c0 + n;
            float v = acc[j][i] * preScale;
            if (bias) v += bias[gc];
            if (actGelu) v = gelu_f(v);
            if (resid) v += resid[(size_t)gr * N + gc];
            if (outF) outF[storeT ? ((size_t)gc * M + gr) : ((size_t)gr * N + gc)] = v;
            if (outH) outH[(size_t)gr * N + gc] = (half_t)v;
        }
    }
}

// ---------------------------------------------------------------------------
// Attention core: one block per (window l in 0..63, head in 0..3).
// Q=384, Kdim=288, DH=32. qh/kh/vh are f16 [L*{384|288},128] row-major.
// ---------------------------------------------------------------------------
__global__ __launch_bounds__(256)
void k_attn(const half_t* __restrict__ qh, const half_t* __restrict__ kh,
            const half_t* __restrict__ vh, half_t* __restrict__ aout)
{
    __shared__ half_t Vt[32][296];   // V transposed: [DH, K]
    __shared__ half_t S[32][296];    // score tile (f16 after scaling / softmax)
    const int l  = blockIdx.x;
    const int hd = blockIdx.y;
    const int t = threadIdx.x;
    const int w = t >> 5, lane = t & 31;
    const float scale = 0.17677669529663687f;  // DH^-0.5

    for (int idx = t; idx < 288 * 32; idx += 256) {
        const int k = idx >> 5, n = idx & 31;
        Vt[n][k] = vh[((size_t)(l * 288 + k)) * 128 + hd * 32 + n];
    }
    __syncthreads();

    for (int qt = 0; qt < 12; ++qt) {
        const int q0 = qt * 32;
        // ---- S = scale * Q K^T ----
        {
            const int mt = w >> 2;                       // 0..1
            const int m = mt * 16 + (lane & 15);
            const unsigned int* arow =
                (const unsigned int*)(qh + ((size_t)(l * 384 + q0 + m)) * 128 + hd * 32);
            v16h af = load_fragA(arow, 0, lane);
            for (int nt = (w & 3); nt < 18; nt += 4) {
                const int n = nt * 16 + (lane & 15);
                const unsigned int* brow =
                    (const unsigned int*)(kh + ((size_t)(l * 288 + n)) * 128 + hd * 32);
                v16h bf = load_fragB(brow, 0, lane);
                v8f acc;
#pragma unroll
                for (int i = 0; i < 8; ++i) acc[i] = 0.0f;
                acc = __builtin_amdgcn_wmma_f32_16x16x32_f16(false, af, false, bf,
                                                             (short)0, acc, false, false);
#pragma unroll
                for (int i = 0; i < 8; ++i) {
                    const int sm = mt * 16 + ((lane >> 4) & 1) * 8 + i;
                    const int sn = nt * 16 + (lane & 15);
                    S[sm][sn] = (half_t)(acc[i] * scale);
                }
            }
        }
        __syncthreads();
        // ---- softmax over 288 cols, 8 threads per row ----
        {
            const int r = t >> 3, g = t & 7;
            float mx = -1e30f;
            for (int c = g * 36; c < g * 36 + 36; ++c) mx = fmaxf(mx, (float)S[r][c]);
#pragma unroll
            for (int msk = 1; msk < 8; msk <<= 1) mx = fmaxf(mx, __shfl_xor(mx, msk, 32));
            float sum = 0.0f;
            for (int c = g * 36; c < g * 36 + 36; ++c) {
                const float e = expf((float)S[r][c] - mx);
                sum += e;
                S[r][c] = (half_t)e;
            }
#pragma unroll
            for (int msk = 1; msk < 8; msk <<= 1) sum += __shfl_xor(sum, msk, 32);
            const float inv = 1.0f / sum;
            for (int c = g * 36; c < g * 36 + 36; ++c) S[r][c] = (half_t)((float)S[r][c] * inv);
        }
        __syncthreads();
        // ---- O = P @ V ----
        if (w < 4) {
            const int mt = w >> 1, nt = w & 1;
            v8f acc;
#pragma unroll
            for (int i = 0; i < 8; ++i) acc[i] = 0.0f;
            const unsigned int* arow = (const unsigned int*)(&S[mt * 16 + (lane & 15)][0]);
            const unsigned int* brow = (const unsigned int*)(&Vt[nt * 16 + (lane & 15)][0]);
#pragma unroll
            for (int ks = 0; ks < 9; ++ks) {
                v16h af = load_fragA(arow, ks * 32, lane);
                v16h bf = load_fragB(brow, ks * 32, lane);
                acc = __builtin_amdgcn_wmma_f32_16x16x32_f16(false, af, false, bf,
                                                             (short)0, acc, false, false);
            }
#pragma unroll
            for (int i = 0; i < 8; ++i) {
                const int m = mt * 16 + ((lane >> 4) & 1) * 8 + i;
                const int n = nt * 16 + (lane & 15);
                aout[((size_t)(l * 384 + q0 + m)) * 128 + hd * 32 + n] = (half_t)acc[i];
            }
        }
        __syncthreads();
    }
}

// ---------------------------------------------------------------------------
// Geometry: img_emb [N=6, P=3072, D=128] row-major (pixel-major, d contiguous)
// wave per (n,p); each lane handles 4 d's.
// ---------------------------------------------------------------------------
__global__ __launch_bounds__(256)
void k_geom(const float* __restrict__ I_inv, const float* __restrict__ E_inv,
            const float* __restrict__ ie_w, const float* __restrict__ ce_w,
            float* __restrict__ imgEmb)
{
    const int idx = blockIdx.x * 8 + (threadIdx.x >> 5);   // n*3072 + p
    const int lane = threadIdx.x & 31;
    const int n = idx / 3072, p = idx % 3072;
    const int i = p / 96, j = p % 96;
    const float px = (float)j * (768.0f / 95.0f);
    const float py = (float)i * (256.0f / 31.0f);
    const float* Ii = I_inv + n * 9;
    const float c0 = Ii[0] * px + Ii[1] * py + Ii[2];
    const float c1 = Ii[3] * px + Ii[4] * py + Ii[5];
    const float c2 = Ii[6] * px + Ii[7] * py + Ii[8];
    const float* Ei = E_inv + n * 16;
    float dv[4];
#pragma unroll
    for (int r = 0; r < 4; ++r)
        dv[r] = Ei[r * 4 + 0] * c0 + Ei[r * 4 + 1] * c1 + Ei[r * 4 + 2] * c2 + Ei[r * 4 + 3];
    float vals[4]; float ss = 0.0f;
#pragma unroll
    for (int u = 0; u < 4; ++u) {
        const int d = lane * 4 + u;
        const float* iw = ie_w + d * 4;
        const float de = iw[0] * dv[0] + iw[1] * dv[1] + iw[2] * dv[2] + iw[3] * dv[3];
        const float* cw = ce_w + d * 4;
        const float ce = cw[0] * Ei[3] + cw[1] * Ei[7] + cw[2] * Ei[11] + cw[3] * Ei[15];
        const float v = de - ce;
        vals[u] = v; ss += v * v;
    }
    ss = wave_sum(ss);
    const float inv = 1.0f / (sqrtf(ss) + 1e-7f);
    float4 o = {vals[0] * inv, vals[1] * inv, vals[2] * inv, vals[3] * inv};
    *(float4*)(imgEmb + (size_t)idx * 128 + lane * 4) = o;
}

// query[N, H*W, D] = normalize(w_embed - c_embed) + x
__global__ __launch_bounds__(256)
void k_query(const float* __restrict__ xin, const float* __restrict__ bev,
             const float* __restrict__ be_w, const float* __restrict__ be_b,
             const float* __restrict__ E_inv, const float* __restrict__ ce_w,
             float* __restrict__ query)
{
    const int idx = blockIdx.x * 8 + (threadIdx.x >> 5);   // n*4096 + hw
    const int lane = threadIdx.x & 31;
    const int n = idx >> 12, hw = idx & 4095;
    const float g0 = bev[hw], g1 = bev[4096 + hw];
    const float* Ei = E_inv + n * 16;
    float vals[4]; float ss = 0.0f;
#pragma unroll
    for (int u = 0; u < 4; ++u) {
        const int d = lane * 4 + u;
        const float we = g0 * be_w[d * 2] + g1 * be_w[d * 2 + 1] + be_b[d];
        const float* cw = ce_w + d * 4;
        const float ce = cw[0] * Ei[3] + cw[1] * Ei[7] + cw[2] * Ei[11] + cw[3] * Ei[15];
        const float v = we - ce;
        vals[u] = v; ss += v * v;
    }
    ss = wave_sum(ss);
    const float inv = 1.0f / (sqrtf(ss) + 1e-7f);
    float4 o;
    o.x = vals[0] * inv + xin[(size_t)(lane * 4 + 0) * 4096 + hw];
    o.y = vals[1] * inv + xin[(size_t)(lane * 4 + 1) * 4096 + hw];
    o.z = vals[2] * inv + xin[(size_t)(lane * 4 + 2) * 4096 + hw];
    o.w = vals[3] * inv + xin[(size_t)(lane * 4 + 3) * 4096 + hw];
    *(float4*)(query + (size_t)idx * 128 + lane * 4) = o;
}

// relu(feature * BN_S) transposed to A16 [N*P, C] f16
__global__ void k_prep_f(const float* __restrict__ feat, half_t* __restrict__ A16)
{
    const int idx = blockIdx.x * 256 + threadIdx.x;   // over 6*128*3072
    const int n = idx / (128 * 3072);
    const int rem = idx % (128 * 3072);
    const int c = rem / 3072, p = rem % 3072;
    float v = feat[idx] * BN_S;
    v = v > 0.0f ? v : 0.0f;
    A16[((size_t)(n * 3072 + p)) * 128 + c] = (half_t)v;
}

// gather + LayerNorm for attention queries -> f16 [64*384, 128]
__global__ __launch_bounds__(256)
void k_ln_gather_q(const float* __restrict__ src, half_t* __restrict__ dst, int mode)
{
    const int row = blockIdx.x * 8 + (threadIdx.x >> 5);   // 24576 rows
    const int lane = threadIdx.x & 31;
    const int l = row / 384, q = row % 384;
    const int n = q >> 6, w1 = (q >> 3) & 7, w2 = q & 7;
    const int x = l >> 3, y = l & 7;
    size_t srcRow;
    if (mode == 0) { const int h = x * 8 + w1, wv = y * 8 + w2; srcRow = (size_t)(n * 4096 + h * 64 + wv); }
    else           { const int h = w1 * 8 + x, wv = w2 * 8 + y; srcRow = (size_t)(h * 64 + wv); }
    const float4 f4 = *(const float4*)(src + srcRow * 128 + lane * 4);
    const float mu = wave_sum(f4.x + f4.y + f4.z + f4.w) * (1.0f / 128.0f);
    const float d0 = f4.x - mu, d1 = f4.y - mu, d2 = f4.z - mu, d3 = f4.w - mu;
    const float var = wave_sum(d0 * d0 + d1 * d1 + d2 * d2 + d3 * d3) * (1.0f / 128.0f);
    const float rs = rsqrtf(var + 1e-5f);
    half_t* o = dst + (size_t)row * 128 + lane * 4;
    o[0] = (half_t)(d0 * rs); o[1] = (half_t)(d1 * rs);
    o[2] = (half_t)(d2 * rs); o[3] = (half_t)(d3 * rs);
}

// gather + LayerNorm for attention keys/values -> f16 [64*288, 128]
__global__ __launch_bounds__(256)
void k_ln_gather_kv(const float* __restrict__ src, half_t* __restrict__ dst, int mode)
{
    const int row = blockIdx.x * 8 + (threadIdx.x >> 5);   // 18432 rows
    const int lane = threadIdx.x & 31;
    const int l = row / 288, kq = row % 288;
    const int n = kq / 48, rem = kq % 48;
    const int k1 = rem / 12, k2 = rem % 12;
    const int x = l >> 3, y = l & 7;
    int h, wv;
    if (mode == 0) { h = x * 4 + k1; wv = y * 12 + k2; }
    else           { h = k1 * 8 + x; wv = k2 * 8 + y; }
    const size_t srcRow = (size_t)(n * 3072 + h * 96 + wv);
    const float4 f4 = *(const float4*)(src + srcRow * 128 + lane * 4);
    const float mu = wave_sum(f4.x + f4.y + f4.z + f4.w) * (1.0f / 128.0f);
    const float d0 = f4.x - mu, d1 = f4.y - mu, d2 = f4.z - mu, d3 = f4.w - mu;
    const float var = wave_sum(d0 * d0 + d1 * d1 + d2 * d2 + d3 * d3) * (1.0f / 128.0f);
    const float rs = rsqrtf(var + 1e-5f);
    half_t* o = dst + (size_t)row * 128 + lane * 4;
    o[0] = (half_t)(d0 * rs); o[1] = (half_t)(d1 * rs);
    o[2] = (half_t)(d2 * rs); o[3] = (half_t)(d3 * rs);
}

// rows [M,128]: optional LayerNorm, write f16 (and optional f32 copy)
__global__ __launch_bounds__(256)
void k_rows_cvt(const float* __restrict__ src, half_t* __restrict__ dst16,
                float* __restrict__ dst32, int doLN)
{
    const int row = blockIdx.x * 8 + (threadIdx.x >> 5);
    const int lane = threadIdx.x & 31;
    const float4 f4 = *(const float4*)(src + (size_t)row * 128 + lane * 4);
    float v[4] = {f4.x, f4.y, f4.z, f4.w};
    if (doLN) {
        const float mu = wave_sum(v[0] + v[1] + v[2] + v[3]) * (1.0f / 128.0f);
        float ss = 0.0f;
#pragma unroll
        for (int i = 0; i < 4; ++i) { v[i] -= mu; ss += v[i] * v[i]; }
        const float rs = rsqrtf(wave_sum(ss) * (1.0f / 128.0f) + 1e-5f);
#pragma unroll
        for (int i = 0; i < 4; ++i) v[i] *= rs;
    }
    half_t* o = dst16 + (size_t)row * 128 + lane * 4;
#pragma unroll
    for (int i = 0; i < 4; ++i) o[i] = (half_t)v[i];
    if (dst32) {
        float4 r = {v[0], v[1], v[2], v[3]};
        *(float4*)(dst32 + (size_t)row * 128 + lane * 4) = r;
    }
}

// mean over cameras + skip, scattered back to q_img [H*W, D]
__global__ __launch_bounds__(256)
void k_reduce_attn(const float* __restrict__ proj, const float* __restrict__ skipChan,
                   const float* __restrict__ skipRow, float* __restrict__ out, int mode)
{
    const int row = blockIdx.x * 8 + (threadIdx.x >> 5);   // h*64+w, 4096 rows
    const int lane = threadIdx.x & 31;
    const int h = row >> 6, wv = row & 63;
    int x, w1, y, w2;
    if (mode == 0) { x = h >> 3; w1 = h & 7; y = wv >> 3; w2 = wv & 7; }
    else           { x = h & 7; w1 = h >> 3; y = wv & 7; w2 = wv >> 3; }
    const int l = x * 8 + y;
    const int d0 = lane * 4;
    float a0 = 0, a1 = 0, a2 = 0, a3 = 0;
#pragma unroll
    for (int n = 0; n < 6; ++n) {
        const int q = n * 64 + w1 * 8 + w2;
        const float4 f = *(const float4*)(proj + ((size_t)(l * 384 + q)) * 128 + d0);
        a0 += f.x; a1 += f.y; a2 += f.z; a3 += f.w;
    }
    const float inv6 = 1.0f / 6.0f;
    a0 *= inv6; a1 *= inv6; a2 *= inv6; a3 *= inv6;
    if (mode == 0) {
        a0 += skipChan[(size_t)(d0 + 0) * 4096 + row];
        a1 += skipChan[(size_t)(d0 + 1) * 4096 + row];
        a2 += skipChan[(size_t)(d0 + 2) * 4096 + row];
        a3 += skipChan[(size_t)(d0 + 3) * 4096 + row];
    } else {
        const float4 s = *(const float4*)(skipRow + (size_t)row * 128 + d0);
        a0 += s.x; a1 += s.y; a2 += s.z; a3 += s.w;
    }
    float4 r = {a0, a1, a2, a3};
    *(float4*)(out + (size_t)row * 128 + d0) = r;
}

// depthwise 3x3 SAME over [H*W, 512] f16 rows; out = gelu(acc * BN_S)
__global__ void k_dwconv(const half_t* __restrict__ ex, const float* __restrict__ dwW,
                         half_t* __restrict__ out)
{
    const int idx = blockIdx.x * 256 + threadIdx.x;   // 4096*512
    const int c = idx & 511, r = idx >> 9;
    const int h = r >> 6, wv = r & 63;
    float acc = 0.0f;
#pragma unroll
    for (int dh = -1; dh <= 1; ++dh)
#pragma unroll
        for (int dw = -1; dw <= 1; ++dw) {
            const int hh = h + dh, ww = wv + dw;
            if (hh < 0 || hh > 63 || ww < 0 || ww > 63) continue;
            acc += (float)ex[((size_t)(hh * 64 + ww)) * 512 + c] * dwW[c * 9 + (dh + 1) * 3 + (dw + 1)];
        }
    out[idx] = (half_t)gelu_f(acc * BN_S);
}

__global__ __launch_bounds__(256)
void k_colmean(const half_t* __restrict__ x, float* __restrict__ mean)
{
    __shared__ float sh[256];
    const int c = blockIdx.x, t = threadIdx.x;
    float s = 0.0f;
    for (int r = t; r < 4096; r += 256) s += (float)x[(size_t)r * 512 + c];
    sh[t] = s; __syncthreads();
    for (int o = 128; o > 0; o >>= 1) { if (t < o) sh[t] += sh[t + o]; __syncthreads(); }
    if (t == 0) mean[c] = sh[0] * (1.0f / 4096.0f);
}

__global__ __launch_bounds__(256)
void k_se(const float* __restrict__ mean, const float* __restrict__ s1w,
          const float* __restrict__ s1b, const float* __restrict__ s2w,
          const float* __restrict__ s2b, float* __restrict__ se)
{
    __shared__ float se1[32];
    const int t = threadIdx.x;
    const int j = t >> 3, g = t & 7;
    float s = 0.0f;
    for (int c = g * 64; c < g * 64 + 64; ++c) s += s1w[j * 512 + c] * mean[c];
#pragma unroll
    for (int m = 1; m < 8; m <<= 1) s += __shfl_xor(s, m, 32);
    if (g == 0) se1[j] = gelu_f(s + s1b[j]);
    __syncthreads();
    for (int c = t; c < 512; c += 256) {
        float d = 0.0f;
#pragma unroll
        for (int k2 = 0; k2 < 32; ++k2) d += s2w[c * 32 + k2] * se1[k2];
        se[c] = 1.0f / (1.0f + expf(-(d + s2b[c])));
    }
}

__global__ void k_scale(const half_t* __restrict__ x, const float* __restrict__ se,
                        half_t* __restrict__ out)
{
    const int i = blockIdx.x * 256 + threadIdx.x;
    out[i] = (half_t)((float)x[i] * se[i & 511]);
}

__global__ void k_cvt16(const float* __restrict__ src, half_t* __restrict__ dst, int nElem)
{
    const int i = blockIdx.x * 256 + threadIdx.x;
    if (i < nElem) dst[i] = (half_t)src[i];
}

// ---------------------------------------------------------------------------

extern "C" void kernel_launch(void* const* d_in, const int* in_sizes, int n_in,
                              void* d_out, int out_size, void* d_ws, size_t ws_size,
                              hipStream_t stream)
{
    (void)in_sizes; (void)n_in; (void)out_size; (void)ws_size;
    const float* x_in  = (const float*)d_in[1];
    const float* feat  = (const float*)d_in[2];
    const float* I_inv = (const float*)d_in[3];
    const float* E_inv = (const float*)d_in[4];
    const float* bev   = (const float*)d_in[5];
    const float* fl_w  = (const float*)d_in[6];
    const float* fp_w  = (const float*)d_in[7];
    const float* be_w  = (const float*)d_in[8];
    const float* be_b  = (const float*)d_in[9];
    const float* ie_w  = (const float*)d_in[10];
    const float* ce_w  = (const float*)d_in[11];
    const float* a1_qw = (const float*)d_in[12]; const float* a1_qb = (const float*)d_in[13];
    const float* a1_kw = (const float*)d_in[14]; const float* a1_kb = (const float*)d_in[15];
    const float* a1_vw = (const float*)d_in[16]; const float* a1_vb = (const float*)d_in[17];
    const float* a1_pw = (const float*)d_in[18]; const float* a1_pb = (const float*)d_in[19];
    const float* a2_qw = (const float*)d_in[20]; const float* a2_qb = (const float*)d_in[21];
    const float* a2_kw = (const float*)d_in[22]; const float* a2_kb = (const float*)d_in[23];
    const float* a2_vw = (const float*)d_in[24]; const float* a2_vb = (const float*)d_in[25];
    const float* a2_pw = (const float*)d_in[26]; const float* a2_pb = (const float*)d_in[27];
    const float* m1_w1 = (const float*)d_in[28]; const float* m1_b1 = (const float*)d_in[29];
    const float* m1_w2 = (const float*)d_in[30]; const float* m1_b2 = (const float*)d_in[31];
    const float* m2_w1 = (const float*)d_in[32]; const float* m2_b1 = (const float*)d_in[33];
    const float* m2_w2 = (const float*)d_in[34]; const float* m2_b2 = (const float*)d_in[35];
    const float* mb_ew = (const float*)d_in[36]; const float* mb_dw = (const float*)d_in[37];
    const float* mb_s1w = (const float*)d_in[38]; const float* mb_s1b = (const float*)d_in[39];
    const float* mb_s2w = (const float*)d_in[40]; const float* mb_s2b = (const float*)d_in[41];
    const float* mb_pw = (const float*)d_in[42];
    const float* pm_w1 = (const float*)d_in[43]; const float* pm_b1 = (const float*)d_in[44];
    const float* pm_w2 = (const float*)d_in[45]; const float* pm_b2 = (const float*)d_in[46];
    const float* po_w  = (const float*)d_in[47]; const float* po_b  = (const float*)d_in[48];

    char* ws = (char*)d_ws;
    size_t off = 0;
    auto alloc = [&](size_t bytes) -> void* {
        void* p = ws + off;
        off += (bytes + 255) & ~(size_t)255;
        return p;
    };

    float*  imgEmb = (float*) alloc(6UL * 3072 * 128 * 4);
    float*  query  = (float*) alloc(6UL * 4096 * 128 * 4);
    half_t* A16    = (half_t*)alloc(18432UL * 128 * 2);
    float*  key    = (float*) alloc(18432UL * 128 * 4);
    float*  val    = (float*) alloc(18432UL * 128 * 4);
    half_t* qln    = (half_t*)alloc(24576UL * 128 * 2);
    half_t* kln    = (half_t*)alloc(18432UL * 128 * 2);
    half_t* vln    = (half_t*)alloc(18432UL * 128 * 2);
    half_t* qhB    = (half_t*)alloc(24576UL * 128 * 2);
    half_t* khB    = (half_t*)alloc(18432UL * 128 * 2);
    half_t* vhB    = (half_t*)alloc(18432UL * 128 * 2);
    half_t* aoutB  = (half_t*)alloc(24576UL * 128 * 2);
    float*  proj   = (float*) alloc(24576UL * 128 * 4);
    float*  qimgA  = (float*) alloc(4096UL * 128 * 4);
    float*  qimgB  = (float*) alloc(4096UL * 128 * 4);
    float*  z32    = (float*) alloc(4096UL * 128 * 4);
    half_t* rows16 = (half_t*)alloc(4096UL * 128 * 2);
    half_t* hid16  = (half_t*)alloc(4096UL * 256 * 2);
    half_t* ex16   = (half_t*)alloc(4096UL * 512 * 2);
    half_t* dw16   = (half_t*)alloc(4096UL * 512 * 2);
    half_t* sc16   = (half_t*)alloc(4096UL * 512 * 2);
    float*  meanB  = (float*) alloc(512 * 4);
    float*  seB    = (float*) alloc(512 * 4);

    auto cvtW = [&](const float* src, int nElem) -> half_t* {
        half_t* dst = (half_t*)alloc((size_t)nElem * 2);
        k_cvt16<<<(nElem + 255) / 256, 256, 0, stream>>>(src, dst, nElem);
        return dst;
    };
    half_t* w_fl  = cvtW(fl_w, 16384);
    half_t* w_fp  = cvtW(fp_w, 16384);
    half_t* w_a1q = cvtW(a1_qw, 16384); half_t* w_a1k = cvtW(a1_kw, 16384);
    half_t* w_a1v = cvtW(a1_vw, 16384); half_t* w_a1p = cvtW(a1_pw, 16384);
    half_t* w_a2q = cvtW(a2_qw, 16384); half_t* w_a2k = cvtW(a2_kw, 16384);
    half_t* w_a2v = cvtW(a2_vw, 16384); half_t* w_a2p = cvtW(a2_pw, 16384);
    half_t* w_m1a = cvtW(m1_w1, 32768); half_t* w_m1b = cvtW(m1_w2, 32768);
    half_t* w_m2a = cvtW(m2_w1, 32768); half_t* w_m2b = cvtW(m2_w2, 32768);
    half_t* w_mbe = cvtW(mb_ew, 65536); half_t* w_mbp = cvtW(mb_pw, 65536);
    half_t* w_pm1 = cvtW(pm_w1, 32768); half_t* w_pm2 = cvtW(pm_w2, 32768);
    half_t* w_po  = cvtW(po_w, 16384);

    auto gemm = [&](const half_t* A, const half_t* B, const float* bias, const float* resid,
                    float* oF, half_t* oH, int M, int N, int K, float preScale, int act, int st) {
        dim3 g(M / 64, N / 128);
        k_gemm<<<g, 256, 0, stream>>>(A, B, bias, resid, oF, oH, M, N, K, preScale, act, st);
    };

    // ---- embeddings, key/value ----
    k_geom <<<2304, 256, 0, stream>>>(I_inv, E_inv, ie_w, ce_w, imgEmb);
    k_query<<<3072, 256, 0, stream>>>(x_in, bev, be_w, be_b, E_inv, ce_w, query);
    k_prep_f<<<(6 * 128 * 3072) / 256, 256, 0, stream>>>(feat, A16);
    gemm(A16, w_fl, nullptr, nullptr, val, nullptr, 18432, 128, 128, 1.0f, 0, 0);
    gemm(A16, w_fp, nullptr, imgEmb, key, nullptr, 18432, 128, 128, 1.0f, 0, 0);

    // ---- attention 1 (local windows) ----
    k_ln_gather_q <<<3072, 256, 0, stream>>>(query, qln, 0);
    k_ln_gather_kv<<<2304, 256, 0, stream>>>(key, kln, 0);
    k_ln_gather_kv<<<2304, 256, 0, stream>>>(val, vln, 0);
    gemm(qln, w_a1q, a1_qb, nullptr, nullptr, qhB, 24576, 128, 128, 1.0f, 0, 0);
    gemm(kln, w_a1k, a1_kb, nullptr, nullptr, khB, 18432, 128, 128, 1.0f, 0, 0);
    gemm(vln, w_a1v, a1_vb, nullptr, nullptr, vhB, 18432, 128, 128, 1.0f, 0, 0);
    k_attn<<<dim3(64, 4), 256, 0, stream>>>(qhB, khB, vhB, aoutB);
    gemm(aoutB, w_a1p, a1_pb, nullptr, proj, nullptr, 24576, 128, 128, 1.0f, 0, 0);
    k_reduce_attn<<<512, 256, 0, stream>>>(proj, x_in, nullptr, qimgA, 0);
    k_rows_cvt<<<512, 256, 0, stream>>>(qimgA, rows16, nullptr, 1);
    gemm(rows16, w_m1a, m1_b1, nullptr, nullptr, hid16, 4096, 256, 128, 1.0f, 1, 0);
    gemm(hid16, w_m1b, m1_b2, qimgA, qimgB, nullptr, 4096, 128, 256, 1.0f, 0, 0);

    // ---- attention 2 (grid windows) ----
    k_ln_gather_q <<<3072, 256, 0, stream>>>(qimgB, qln, 1);
    k_ln_gather_kv<<<2304, 256, 0, stream>>>(key, kln, 1);
    k_ln_gather_kv<<<2304, 256, 0, stream>>>(val, vln, 1);
    gemm(qln, w_a2q, a2_qb, nullptr, nullptr, qhB, 24576, 128, 128, 1.0f, 0, 0);
    gemm(kln, w_a2k, a2_kb, nullptr, nullptr, khB, 18432, 128, 128, 1.0f, 0, 0);
    gemm(vln, w_a2v, a2_vb, nullptr, nullptr, vhB, 18432, 128, 128, 1.0f, 0, 0);
    k_attn<<<dim3(64, 4), 256, 0, stream>>>(qhB, khB, vhB, aoutB);
    gemm(aoutB, w_a2p, a2_pb, nullptr, proj, nullptr, 24576, 128, 128, 1.0f, 0, 0);
    k_reduce_attn<<<512, 256, 0, stream>>>(proj, nullptr, qimgB, qimgA, 1);
    k_rows_cvt<<<512, 256, 0, stream>>>(qimgA, rows16, nullptr, 1);
    gemm(rows16, w_m2a, m2_b1, nullptr, nullptr, hid16, 4096, 256, 128, 1.0f, 1, 0);
    gemm(hid16, w_m2b, m2_b2, qimgA, qimgB, nullptr, 4096, 128, 256, 1.0f, 0, 0);

    // ---- postnorm + MBConv ----
    k_rows_cvt<<<512, 256, 0, stream>>>(qimgB, rows16, z32, 1);
    gemm(rows16, w_mbe, nullptr, nullptr, nullptr, ex16, 4096, 512, 128, BN_S, 1, 0);
    k_dwconv<<<8192, 256, 0, stream>>>(ex16, mb_dw, dw16);
    k_colmean<<<512, 256, 0, stream>>>(dw16, meanB);
    k_se<<<1, 256, 0, stream>>>(meanB, mb_s1w, mb_s1b, mb_s2w, mb_s2b, seB);
    k_scale<<<8192, 256, 0, stream>>>(dw16, seB, sc16);
    gemm(sc16, w_mbp, nullptr, z32, qimgA, nullptr, 4096, 128, 512, BN_S, 0, 0);

    // ---- channel FFN + proj_out ----
    k_rows_cvt<<<512, 256, 0, stream>>>(qimgA, rows16, nullptr, 0);
    gemm(rows16, w_pm1, pm_b1, nullptr, nullptr, hid16, 4096, 256, 128, 1.0f, 1, 0);
    gemm(hid16, w_pm2, pm_b2, qimgA, qimgB, nullptr, 4096, 128, 256, 1.0f, 0, 0);
    k_rows_cvt<<<512, 256, 0, stream>>>(qimgB, rows16, nullptr, 0);
    gemm(rows16, w_po, po_b, nullptr, (float*)d_out, nullptr, 4096, 128, 128, 1.0f, 1, 1);
}
